// ContrastiveLoss_2027224563777
// MI455X (gfx1250) — matrix operand order repaired and checked by reading, not compiled, not run on hardware
//
#include <hip/hip_runtime.h>
#include <hip/hip_bf16.h>

// ---------------------------------------------------------------------------
// NT-Xent contrastive loss, fused for MI455X (gfx1250, wave32, WMMA).
//   Z = f16(normalize(concat(proj_1, proj_2)))            [8192 x 128]
//   sim tiles (16x16) via v_wmma_f32_16x16x32_f16, K=128 in 4 chunks
//   Fused exp/row-sum, sim never materialized (saves 2x256MB of traffic).
//   v3: sched_barrier-enforced register double buffering (s_wait_loadcnt 8
//       instead of 0), exp2-folded temperature (1 mul + 1 trans per elem),
//       global_prefetch_b8 two tiles ahead.
// ---------------------------------------------------------------------------

typedef __attribute__((ext_vector_type(16))) _Float16 v16h;
typedef __attribute__((ext_vector_type(8)))  float    v8f;
typedef __attribute__((ext_vector_type(4)))  float    v4f;
typedef __attribute__((ext_vector_type(4)))  unsigned int v4u;

#define NROWS   4096
#define DIM     128
#define TWO_N   8192
#define NTILES  512        // TWO_N / 16
#define INV_T   2.0f       // 1 / TEMPERATURE
#define EXP2_K  2.8853900817779268f   // (1/T) * log2(e): exp(x/T)=exp2(x*K)
#define WPB     8          // waves per block (column split factor)
#define TPW     (NTILES / WPB)   // 64 column tiles per wave

// ---- Kernel 1: L2 normalize, f32 -> f16, one wave32 per row ---------------
__global__ __launch_bounds__(256)
void ntx_normalize(const float* __restrict__ p1, const float* __restrict__ p2,
                   _Float16* __restrict__ Z) {
  const int wave = (blockIdx.x * blockDim.x + threadIdx.x) >> 5;  // row id
  const int lane = threadIdx.x & 31;
  const float* src = (wave < NROWS) ? (p1 + (size_t)wave * DIM)
                                    : (p2 + (size_t)(wave - NROWS) * DIM);
  v4f v = *(const v4f*)(src + lane * 4);                 // 4 floats / lane
  float s = v.x * v.x + v.y * v.y + v.z * v.z + v.w * v.w;
  #pragma unroll
  for (int m = 1; m < 32; m <<= 1) s += __shfl_xor(s, m, 32);
  const float scale = 1.0f / fmaxf(sqrtf(s), 1e-12f);
  union { _Float16 h[4]; unsigned long long u; } out;
  out.h[0] = (_Float16)(v.x * scale);
  out.h[1] = (_Float16)(v.y * scale);
  out.h[2] = (_Float16)(v.z * scale);
  out.h[3] = (_Float16)(v.w * scale);
  *(unsigned long long*)(Z + (size_t)wave * DIM + lane * 4) = out.u;
}

// ---- B fragment: 32x16 f16, K split into 4 chunks of 32 -------------------
struct Bfrag { v4u q[8]; };   // chunk c lives in q[2c], q[2c+1]

__device__ __forceinline__
void load_b(Bfrag& B, const _Float16* __restrict__ Z, int colBase,
            int m16, int half) {
  // B layout: lane n = l%16; elem j -> K = 32c + (j%8) + half*8 + (j/8)*16.
  const _Float16* bp = Z + (size_t)(colBase + m16) * DIM + half * 8;
  #pragma unroll
  for (int c = 0; c < 4; ++c) {
    B.q[2 * c]     = *(const v4u*)(bp + 32 * c);
    B.q[2 * c + 1] = *(const v4u*)(bp + 32 * c + 16);
  }
}

__device__ __forceinline__
v8f mm_tile(const v16h A[4], const Bfrag& B) {
  v8f c8 = {};
  #pragma unroll
  for (int c = 0; c < 4; ++c) {
    union { v4u q[2]; v16h v; } bb;
    bb.q[0] = B.q[2 * c];
    bb.q[1] = B.q[2 * c + 1];
    c8 = __builtin_amdgcn_wmma_f32_16x16x32_f16(
        /*neg_a=*/false, A[c], /*neg_b=*/false, bb.v,
        /*c_mod=*/(short)0, c8, /*reuse_a=*/false, /*reuse_b=*/false);
  }
  return c8;
}

__device__ __forceinline__
void acc_exp(float acc[8], const v8f& c8) {
  // exp(s/T) as a single v_mul + v_exp_f32 (temperature folded into log2e).
  #pragma unroll
  for (int r = 0; r < 8; ++r)
    acc[r] += __builtin_amdgcn_exp2f(c8[r] * EXP2_K);
}

// ---- Kernel 2: fused sim + exp + row-sum --------------------------------
// One block (8 waves) per 16-row tile; each wave sweeps 64 column tiles.
__global__ __launch_bounds__(256)
void ntx_main(const _Float16* __restrict__ Z, float* __restrict__ partial) {
  const int lane    = threadIdx.x & 31;
  const int m16     = lane & 15;       // D-layout: n = lane % 16
  const int half    = lane >> 4;       // D-layout: m = r + (lane/16)*8
  const int waveId  = threadIdx.x >> 5;
  const int rowTile = blockIdx.x;      // 0..511
  const int rowBase = rowTile * 16;
  const int tBeg    = waveId * TPW;    // this wave's column-tile range

  // Preload A fragments for all 4 K-chunks (reused across all tiles).
  // A layout: lane m = l%16; elem j -> K = 32c + (j%8) + half*8 + (j/8)*16.
  union Af { v16h v; unsigned int u[8]; };
  Af A[4];
  const _Float16* arow = Z + (size_t)(rowBase + m16) * DIM;
  #pragma unroll
  for (int c = 0; c < 4; ++c) {
    #pragma unroll
    for (int t = 0; t < 8; ++t) {
      const int k = 32 * c + half * 8 + ((t < 4) ? (2 * t) : (2 * (t - 4) + 16));
      A[c].u[t] = *(const unsigned int*)(arow + k);       // 2 packed halves
    }
  }
  v16h Av[4] = { A[0].v, A[1].v, A[2].v, A[3].v };

  float acc[8];
  #pragma unroll
  for (int r = 0; r < 8; ++r) acc[r] = 0.0f;

  // ---- hot loop: mask-free, sched_barrier-pinned double buffering --------
  // Phase order per iteration (fenced so the scheduler cannot re-fuse the
  // buffers and re-serialize into s_wait_loadcnt 0):
  //   [loads t+1] | [wmma+exp t] | [loads t+2] | [wmma+exp t+1]
  Bfrag B0, B1;
  load_b(B0, Z, tBeg * 16, m16, half);
  __builtin_amdgcn_sched_barrier(0);
  #pragma unroll 1
  for (int t = 0; t < TPW; t += 2) {
    load_b(B1, Z, (tBeg + t + 1) * 16, m16, half);        // prefetch t+1
    __builtin_amdgcn_sched_barrier(0);
    acc_exp(acc, mm_tile(Av, B0));                        // compute t
    __builtin_amdgcn_sched_barrier(0);
    const int t2 = (t + 2 < TPW) ? (t + 2) : (TPW - 1);   // clamp (dup ok)
    load_b(B0, Z, (tBeg + t2) * 16, m16, half);           // prefetch t+2
    // L0 warm-up two tiles ahead of the register pipeline.
    const int t3 = (t + 3 < TPW) ? (t + 3) : (TPW - 1);
    __builtin_prefetch(Z + (size_t)((tBeg + t3) * 16 + m16) * DIM, 0, 3);
    __builtin_amdgcn_sched_barrier(0);
    acc_exp(acc, mm_tile(Av, B1));                        // compute t+1
    __builtin_amdgcn_sched_barrier(0);
  }

  // ---- corrections (wave-uniform branches; EXEC stays all-ones) ----------
  float posv[8];
  #pragma unroll
  for (int r = 0; r < 8; ++r) posv[r] = 0.0f;

  if (rowTile >= tBeg && rowTile < tBeg + TPW) {          // diagonal owner
    Bfrag Bd;
    load_b(Bd, Z, rowBase, m16, half);
    v8f c8 = mm_tile(Av, Bd);
    #pragma unroll
    for (int r = 0; r < 8; ++r)
      if (m16 == r + half * 8)
        acc[r] -= __builtin_amdgcn_exp2f(c8[r] * EXP2_K);
  }
  const int posTile = (rowTile < 256) ? rowTile + 256 : rowTile - 256;
  if (posTile >= tBeg && posTile < tBeg + TPW) {          // positive owner
    Bfrag Bp;
    load_b(Bp, Z, posTile * 16, m16, half);
    v8f c8 = mm_tile(Av, Bp);
    #pragma unroll
    for (int r = 0; r < 8; ++r)
      if (m16 == r + half * 8) posv[r] = c8[r];
  }

  // ---- intra-wave reduction, then cross-wave combine in LDS --------------
  __shared__ float sd[WPB][16];
  __shared__ float sp[WPB][16];
  #pragma unroll
  for (int r = 0; r < 8; ++r) {
    float d = acc[r];
    float p = posv[r];
    #pragma unroll
    for (int m = 1; m < 16; m <<= 1) {
      d += __shfl_xor(d, m, 32);
      p += __shfl_xor(p, m, 32);
    }
    if (m16 == 0) {                       // lane 0 / lane 16 own 8 rows each
      sd[waveId][r + half * 8] = d;
      sp[waveId][r + half * 8] = p;
    }
  }
  __syncthreads();

  if (threadIdx.x < 16) {                 // 16 rows of this row tile
    const int row = threadIdx.x;
    float d = 0.0f, p = 0.0f;
    #pragma unroll
    for (int w = 0; w < WPB; ++w) { d += sd[w][row]; p += sp[w][row]; }
    float loss = __logf(d) - p * INV_T;   // per-row loss
    #pragma unroll
    for (int m = 1; m < 16; m <<= 1) loss += __shfl_xor(loss, m, 32);
    if (row == 0) partial[rowTile] = loss;
  }
}

// ---- Kernel 3: deterministic tree reduction over 512 partials -------------
__global__ __launch_bounds__(512)
void ntx_finish(const float* __restrict__ partial, float* __restrict__ out) {
  __shared__ float sm[512];
  sm[threadIdx.x] = partial[threadIdx.x];
  __syncthreads();
  #pragma unroll
  for (int s = 256; s > 0; s >>= 1) {
    if (threadIdx.x < (unsigned)s) sm[threadIdx.x] += sm[threadIdx.x + s];
    __syncthreads();
  }
  if (threadIdx.x == 0) out[0] = sm[0] * (1.0f / (float)TWO_N);
}

extern "C" void kernel_launch(void* const* d_in, const int* in_sizes, int n_in,
                              void* d_out, int out_size, void* d_ws, size_t ws_size,
                              hipStream_t stream) {
  (void)in_sizes; (void)n_in; (void)out_size; (void)ws_size;
  const float* p1 = (const float*)d_in[0];
  const float* p2 = (const float*)d_in[1];
  _Float16* Z       = (_Float16*)d_ws;                         // 2 MB
  float*    partial = (float*)((char*)d_ws +
                               (size_t)TWO_N * DIM * sizeof(_Float16));

  ntx_normalize<<<TWO_N / 8, 256, 0, stream>>>(p1, p2, Z);     // 8 waves/blk
  ntx_main     <<<NTILES, 256, 0, stream>>>(Z, partial);       // 8 waves/tile
  ntx_finish   <<<1, 512, 0, stream>>>(partial, (float*)d_out);
}
